// Net_46480136077706
// MI455X (gfx1250) — matrix operand order repaired and checked by reading, not compiled, not run on hardware
//
#include <hip/hip_runtime.h>
#include <hip/hip_bf16.h>
#include <math.h>
#include <stdint.h>

// ---------------- problem constants (from reference) ----------------
#define N_OBJS      128
#define D_OBJ       64
#define D_MODEL     256
#define OTHER_START 64
#define D_OBS       (OTHER_START + N_OBJS * D_OBJ)   // 8256
#define IN_ROW      (D_OBS + N_OBJS)                 // 8384
#define HIDDEN      256
#define OUT_ROW     131                              // value(1) + action(2) + attn(128)
#define STATE_DIM   (OTHER_START + D_MODEL)          // 320

// ---------------- WMMA vector types (gfx1250) ----------------
typedef __attribute__((ext_vector_type(16))) __bf16 v16bf;
typedef __attribute__((ext_vector_type(8)))  __bf16 v8bf;
typedef __attribute__((ext_vector_type(8)))  float  v8f;

union FragA { v16bf v; v8bf h[2]; };

// ---------------- LDS layout (bytes into dynamic smem blob) ----------------
#define OFF_URT     0               // 256*256 bf16 = 131072  (Ur^T, [N][K])
#define OFF_WEMBT   131072          // 256*64  bf16 = 32768   (W_emb^T, [N][K])
#define OFF_XREAL   163840          // 128*256 bf16 = 65536   (x_real, row-major [M][K])
                                    //   (re-used as 32KB f32 async staging for `others`
                                    //    before stage B overwrites it)
#define OFF_OTHERS  229376          // 128*64  bf16 = 16384
#define OFF_MASK    245760          // 128 f32
#define OFF_QUERY   246272          // 256 f32
#define OFF_QUQ     247296          // 256 f32
#define OFF_UA      248320          // 256 f32
#define OFF_BEMB    249344          // 256 f32
#define OFF_LOGITS  250368          // 128 f32
#define OFF_ATTN    250880          // 128 f32
#define OFF_STATE   251392          // 320 f32
#define OFF_H1      252672          // 256 f32
#define OFF_H2      253696          // 256 f32
#define OFF_RED     254720          // small scratch
#define SMEM_BYTES  254784          // < 320 KB per WGP

__global__ __launch_bounds__(256, 1)
void fused_policy_kernel(const float* __restrict__ inputs,
                         const float* __restrict__ W_emb, const float* __restrict__ b_emb,
                         const float* __restrict__ Uq,    const float* __restrict__ Ur,
                         const float* __restrict__ Ua,
                         const float* __restrict__ W1p,   const float* __restrict__ b1p,
                         const float* __restrict__ W2p,   const float* __restrict__ b2p,
                         const float* __restrict__ W3p,   const float* __restrict__ b3p,
                         const float* __restrict__ W1v,   const float* __restrict__ b1v,
                         const float* __restrict__ W2v,   const float* __restrict__ b2v,
                         const float* __restrict__ W3v,   const float* __restrict__ b3v,
                         float* __restrict__ out)
{
    extern __shared__ char smem[];
    __bf16* UrT     = (__bf16*)(smem + OFF_URT);
    __bf16* WembT   = (__bf16*)(smem + OFF_WEMBT);
    __bf16* xreal   = (__bf16*)(smem + OFF_XREAL);
    float*  othF    = (float*)(smem + OFF_XREAL);   // 32KB f32 staging (aliased, pre-stage-B only)
    __bf16* othersB = (__bf16*)(smem + OFF_OTHERS);
    float*  maskS   = (float*)(smem + OFF_MASK);
    float*  queryS  = (float*)(smem + OFF_QUERY);
    float*  qUqS    = (float*)(smem + OFF_QUQ);
    float*  UaS     = (float*)(smem + OFF_UA);
    float*  bembS   = (float*)(smem + OFF_BEMB);
    float*  logitsS = (float*)(smem + OFF_LOGITS);
    float*  attnS   = (float*)(smem + OFF_ATTN);
    float*  stateS  = (float*)(smem + OFF_STATE);
    float*  h1S     = (float*)(smem + OFF_H1);
    float*  h2S     = (float*)(smem + OFF_H2);
    float*  redS    = (float*)(smem + OFF_RED);

    const int t    = threadIdx.x;
    const int b    = blockIdx.x;
    const int lane = t & 31;
    const int wave = t >> 5;
    const int half = lane >> 4;      // K-block selector within fragments
    const int lr   = lane & 15;      // row (A) / col (B,C) within 16

    const float* row    = inputs + (size_t)b * IN_ROW;
    const float* others = row + OTHER_START;     // [128][64] f32
    const float* maskG  = row + D_OBS;           // [128]     f32

    // ---------------- Stage A0: async-copy `others` (32KB f32) into LDS ---------
    // GV mode: per-lane LDS byte address (generic LDS ptr low 32 bits) + 64-bit
    // global address. Overlaps with the weight-conversion VALU work below.
    {
        const uint32_t ldsBase = (uint32_t)(uintptr_t)othF;
#pragma unroll
        for (int it = 0; it < 8; ++it) {
            const int fidx = (t + it * 256) * 4;                 // 4 floats = 16B per lane
            const uint32_t ldsA = ldsBase + (uint32_t)fidx * 4u;
            const uint64_t ga   = (uint64_t)(uintptr_t)(others + fidx);
            asm volatile("global_load_async_to_lds_b128 %0, %1, off"
                         :: "v"(ldsA), "v"(ga) : "memory");
        }
    }

    // ---------------- Stage A1: weights -> LDS bf16 (overlaps async copy) -------
    for (int i = t; i < D_MODEL * D_MODEL; i += 256) {         // Ur (K,N) -> UrT [N][K]
        int k = i >> 8, n = i & 255;
        UrT[n * D_MODEL + k] = (__bf16)Ur[i];
    }
    for (int i = t; i < D_OBJ * D_MODEL; i += 256) {           // W_emb (K=64,N=256) -> WembT [N][K]
        int k = i >> 8, n = i & 255;
        WembT[n * D_OBJ + k] = (__bf16)W_emb[i];
    }
    if (t < N_OBJS) {
        float mk = maskG[t];
        maskS[t]   = mk;
        logitsS[t] = (1.0f - mk) * -1.0e9f;                    // mask penalty pre-seeded
    }
    if (t < D_MODEL) {
        queryS[t] = 0.0f;
        UaS[t]    = Ua[t];
        bembS[t]  = b_emb[t];
    }

    // ---------------- Stage A2: wait for async copy, convert others -> bf16 -----
    asm volatile("s_wait_asynccnt 0x0" ::: "memory");
    __syncthreads();
    for (int i = t; i < N_OBJS * D_OBJ; i += 256)
        othersB[i] = (__bf16)othF[i];
    __syncthreads();   // othersB ready; othF (== xreal region) now dead

    // ---------------- Stage B: x_real = (others @ W_emb + b) * mask  (WMMA) ------
    // wave w owns mtile w (rows m0..m0+15); A fragments cached in registers.
    {
        const int m0 = wave * 16;
        FragA a[2];
#pragma unroll
        for (int kk = 0; kk < 2; ++kk) {
            const __bf16* ap = &othersB[(m0 + lr) * D_OBJ + kk * 32 + half * 16];
            a[kk].h[0] = *(const v8bf*)ap;
            a[kk].h[1] = *(const v8bf*)(ap + 8);
        }
        float mrow[8];
#pragma unroll
        for (int r = 0; r < 8; ++r) mrow[r] = maskS[m0 + r + 8 * half];

        for (int nt = 0; nt < 16; ++nt) {
            const int n0 = nt * 16;
            v8f c = {};
#pragma unroll
            for (int kk = 0; kk < 2; ++kk) {
                v16bf bb = *(const v16bf*)&WembT[(n0 + lr) * D_OBJ + kk * 32 + half * 16];
                c = __builtin_amdgcn_wmma_f32_16x16x32_bf16(false, a[kk].v, false, bb,
                                                            (short)0, c, false, false);
            }
            const float bias = bembS[n0 + lr];
            float q = 0.0f;
#pragma unroll
            for (int r = 0; r < 8; ++r) {
                const int m = m0 + r + 8 * half;
                float v = (c[r] + bias) * mrow[r];
                xreal[m * D_MODEL + n0 + lr] = (__bf16)v;
                q += v;
            }
            atomicAdd(&queryS[n0 + lr], q);                    // ds_add_f32
        }
    }
    __syncthreads();

    // ---------------- Stage C: query scale + qUq --------------------------------
    if (t == 0) {
        float s = 0.0f;
        for (int i = 0; i < N_OBJS; ++i) s += maskS[i];
        redS[0] = 1.0f / (s + 1e-5f);
    }
    __syncthreads();
    if (t < D_MODEL) queryS[t] *= redS[0];
    __syncthreads();
    if (t < D_MODEL) {
        float acc = 0.0f;
        for (int k = 0; k < D_MODEL; ++k) acc += queryS[k] * Uq[k * D_MODEL + t];
        qUqS[t] = acc;
    }
    __syncthreads();

    // ---------------- Stage D: logits = tanh(qUq + x_real@Ur) @ Ua  (WMMA) ------
    // wave w owns mtile w; 8 K-step A fragments held in registers (64 VGPRs).
    {
        const int m0 = wave * 16;
        FragA a[8];
#pragma unroll
        for (int kk = 0; kk < 8; ++kk) {
            const __bf16* ap = &xreal[(m0 + lr) * D_MODEL + kk * 32 + half * 16];
            a[kk].h[0] = *(const v8bf*)ap;
            a[kk].h[1] = *(const v8bf*)(ap + 8);
        }

        for (int nt = 0; nt < 16; ++nt) {
            const int n0 = nt * 16;
            v8f c = {};
#pragma unroll
            for (int kk = 0; kk < 8; ++kk) {
                v16bf bb = *(const v16bf*)&UrT[(n0 + lr) * D_MODEL + kk * 32 + half * 16];
                c = __builtin_amdgcn_wmma_f32_16x16x32_bf16(false, a[kk].v, false, bb,
                                                            (short)0, c, false, false);
            }
            const float addq = qUqS[n0 + lr];
            const float ua   = UaS[n0 + lr];
            float part[8];
#pragma unroll
            for (int r = 0; r < 8; ++r) part[r] = tanhf(c[r] + addq) * ua;
            // butterfly-reduce the 16 lanes of each half (the 16 N columns)
#pragma unroll
            for (int off = 1; off < 16; off <<= 1) {
#pragma unroll
                for (int r = 0; r < 8; ++r) part[r] += __shfl_xor(part[r], off, 32);
            }
            if (lr == 0) {
#pragma unroll
                for (int r = 0; r < 8; ++r)
                    atomicAdd(&logitsS[m0 + r + 8 * half], part[r]);
            }
        }
    }
    __syncthreads();

    // ---------------- Stage E: softmax over 128 objects -------------------------
    if (t == 0) {
        float mx = -3.0e38f;
        for (int i = 0; i < N_OBJS; ++i) mx = fmaxf(mx, logitsS[i]);
        redS[1] = mx;
    }
    __syncthreads();
    if (t < N_OBJS) attnS[t] = __expf(logitsS[t] - redS[1]);
    __syncthreads();
    if (t == 0) {
        float s = 0.0f;
        for (int i = 0; i < N_OBJS; ++i) s += attnS[i];
        redS[2] = 1.0f / s;
    }
    __syncthreads();
    if (t < N_OBJS) {
        attnS[t] *= redS[2];
        out[(size_t)b * OUT_ROW + 3 + t] = attnS[t];
    }
    __syncthreads();

    // ---------------- Stage F: pooled + state -----------------------------------
    // masked rows have attn == 0 exactly, so attn @ x_real == attn @ x_others
    if (t < OTHER_START) stateS[t] = row[t];
    if (t < D_MODEL) {
        float p = 0.0f;
        for (int n = 0; n < N_OBJS; ++n)
            p += attnS[n] * (float)xreal[n * D_MODEL + t];
        stateS[OTHER_START + t] = p;
    }
    __syncthreads();

    // ---------------- Stage G: policy & value MLPs (f32 VALU, tiny) -------------
    {   // policy layer 1: state(320) @ W1p(320,256)
        float h = b1p[t];
        for (int k = 0; k < STATE_DIM; ++k) h += stateS[k] * W1p[k * HIDDEN + t];
        h1S[t] = tanhf(h);
    }
    __syncthreads();
    {   // policy layer 2
        float h = b2p[t];
        for (int k = 0; k < HIDDEN; ++k) h += h1S[k] * W2p[k * HIDDEN + t];
        h2S[t] = tanhf(h);
    }
    __syncthreads();
    if (t < 2) {  // policy head: mean -> action = tanh(mean)
        float h = b3p[t];
        for (int k = 0; k < HIDDEN; ++k) h += h2S[k] * W3p[k * 4 + t];
        out[(size_t)b * OUT_ROW + 1 + t] = tanhf(h);
    }
    {   // value layer 1 (h1S reuse is safe: policy readers of h1S are barriered off)
        float h = b1v[t];
        for (int k = 0; k < STATE_DIM; ++k) h += stateS[k] * W1v[k * HIDDEN + t];
        h1S[t] = tanhf(h);
    }
    __syncthreads();
    {   // value layer 2
        float h = b2v[t];
        for (int k = 0; k < HIDDEN; ++k) h += h1S[k] * W2v[k * HIDDEN + t];
        h2S[t] = tanhf(h);
    }
    __syncthreads();
    if (t == 0) {  // value head
        float h = b3v[0];
        for (int k = 0; k < HIDDEN; ++k) h += h2S[k] * W3v[k];
        out[(size_t)b * OUT_ROW] = h;
    }
}

extern "C" void kernel_launch(void* const* d_in, const int* in_sizes, int n_in,
                              void* d_out, int out_size, void* d_ws, size_t ws_size,
                              hipStream_t stream) {
    (void)n_in; (void)out_size; (void)d_ws; (void)ws_size;
    const float* inputs = (const float*)d_in[0];
    const float* W_emb  = (const float*)d_in[1];
    const float* b_emb  = (const float*)d_in[2];
    const float* Uq     = (const float*)d_in[3];
    const float* Ur     = (const float*)d_in[4];
    const float* Ua     = (const float*)d_in[5];
    const float* W1p    = (const float*)d_in[6];
    const float* b1p    = (const float*)d_in[7];
    const float* W2p    = (const float*)d_in[8];
    const float* b2p    = (const float*)d_in[9];
    const float* W3p    = (const float*)d_in[10];
    const float* b3p    = (const float*)d_in[11];
    const float* W1v    = (const float*)d_in[12];
    const float* b1v    = (const float*)d_in[13];
    const float* W2v    = (const float*)d_in[14];
    const float* b2v    = (const float*)d_in[15];
    const float* W3v    = (const float*)d_in[16];
    const float* b3v    = (const float*)d_in[17];

    const int B = in_sizes[0] / IN_ROW;   // 4096

    hipFuncSetAttribute((const void*)fused_policy_kernel,
                        hipFuncAttributeMaxDynamicSharedMemorySize, SMEM_BYTES);

    fused_policy_kernel<<<B, 256, SMEM_BYTES, stream>>>(
        inputs, W_emb, b_emb, Uq, Ur, Ua,
        W1p, b1p, W2p, b2p, W3p, b3p,
        W1v, b1v, W2v, b2v, W3v, b3v,
        (float*)d_out);
}
